// SoftDTW_80444737454399
// MI455X (gfx1250) — matrix operand order repaired and checked by reading, not compiled, not run on hardware
//
#include <hip/hip_runtime.h>

typedef __attribute__((ext_vector_type(16))) _Float16 v16h;
typedef __attribute__((ext_vector_type(8)))  _Float16 v8h;
typedef __attribute__((ext_vector_type(4)))  _Float16 v4h;
typedef __attribute__((ext_vector_type(8)))  float    v8f;
typedef __attribute__((ext_vector_type(4)))  float    v4f;

static constexpr int NB = 64;    // batch
static constexpr int NN = 512;   // sequence length
static constexpr int ND = 64;    // feature dim

// ---------------------------------------------------------------------------
// Kernel 1: per-row f32 -> f16 conversion + squared-norm. One thread per row.
// rows [0, NB*NN) -> X, rows [NB*NN, 2*NB*NN) -> Y.
// ---------------------------------------------------------------------------
__global__ void softdtw_prep_kernel(const float* __restrict__ X,
                                    const float* __restrict__ Y,
                                    _Float16* __restrict__ Xh,
                                    _Float16* __restrict__ Yh,
                                    float* __restrict__ xsq,
                                    float* __restrict__ ysq) {
  int row = blockIdx.x * blockDim.x + threadIdx.x;
  const int totalRows = NB * NN;
  const float* src;
  _Float16* dsth;
  float* sq;
  int r;
  if (row < totalRows) { src = X; dsth = Xh; sq = xsq; r = row; }
  else                 { src = Y; dsth = Yh; sq = ysq; r = row - totalRows; }

  const float* p = src + (size_t)r * ND;
  _Float16*    q = dsth + (size_t)r * ND;

  float acc = 0.0f;
#pragma unroll
  for (int k = 0; k < ND; k += 4) {
    v4f v = *(const v4f*)(p + k);
    acc += v[0] * v[0] + v[1] * v[1] + v[2] * v[2] + v[3] * v[3];
    v4h h;
    h[0] = (_Float16)v[0];
    h[1] = (_Float16)v[1];
    h[2] = (_Float16)v[2];
    h[3] = (_Float16)v[3];
    *(v4h*)(q + k) = h;
  }
  sq[r] = acc;
}

// ---------------------------------------------------------------------------
// Kernel 2: batched GEMM via WMMA -> Euclidean distance matrix D[b][i][j].
// One wave per 16x16 output tile; K=64 -> 2x v_wmma_f32_16x16x32_f16.
// CDNA5 fragment layouts (ISA 7.12.2):
//   A (16x32 f16): lane = (lh, lr): row M=lr; e<8 -> K=lh*8+e, e>=8 -> K=16+lh*8+(e-8)
//   B (32x16 f16): lane = (lh, lr): col N=lr; e<8 -> K=lh*8+e, e>=8 -> K=16+lh*8+(e-8)
//   C/D (16x16 f32): VGPR r, lane (lh, lr): M=r+lh*8, N=lr
// ---------------------------------------------------------------------------
__global__ void softdtw_gemm_dist_kernel(const _Float16* __restrict__ Xh,
                                         const _Float16* __restrict__ Yh,
                                         const float* __restrict__ xsq,
                                         const float* __restrict__ ysq,
                                         float* __restrict__ Dm) {
  const int wid  = blockIdx.x * 8 + (threadIdx.x >> 5);   // 256 thr = 8 waves
  const int lane = threadIdx.x & 31;
  const int b    = wid >> 10;          // / (32*32)
  const int ti   = (wid >> 5) & 31;    // row tile
  const int tj   = wid & 31;           // col tile
  const int lh   = lane >> 4;          // lane half
  const int lr   = lane & 15;

  const _Float16* xrow = Xh + ((size_t)b * NN + ti * 16 + lr) * ND;
  const _Float16* yrow = Yh + ((size_t)b * NN + tj * 16 + lr) * ND;

  v8f c = {};
#pragma unroll
  for (int k0 = 0; k0 < ND; k0 += 32) {
    v8h a0 = *(const v8h*)(xrow + k0 + lh * 8);
    v8h a1 = *(const v8h*)(xrow + k0 + 16 + lh * 8);
    v8h b0 = *(const v8h*)(yrow + k0 + lh * 8);
    v8h b1 = *(const v8h*)(yrow + k0 + 16 + lh * 8);
    v16h A, Bm;
#pragma unroll
    for (int e = 0; e < 8; ++e) {
      A[e]      = a0[e];
      A[e + 8]  = a1[e];
      Bm[e]     = b0[e];
      Bm[e + 8] = b1[e];
    }
    c = __builtin_amdgcn_wmma_f32_16x16x32_f16(false, A, false, Bm,
                                               (short)0, c, false, false);
  }

  const float* xsp = xsq + (size_t)b * NN + ti * 16 + lh * 8;
  v4f xs0 = *(const v4f*)xsp;
  v4f xs1 = *(const v4f*)(xsp + 4);
  float ys = ysq[(size_t)b * NN + tj * 16 + lr];

  float* drow = Dm + ((size_t)b * NN + ti * 16 + lh * 8) * NN + tj * 16 + lr;
#pragma unroll
  for (int r = 0; r < 8; ++r) {
    float xs = (r < 4) ? xs0[r] : xs1[r - 4];
    float d2 = xs + ys - 2.0f * c[r];
    drow[(size_t)r * NN] = sqrtf(fmaxf(d2, 0.0f));
  }
}

// ---------------------------------------------------------------------------
// Kernel 3: soft-DTW DP over anti-diagonals. One block per batch, thread t
// owns column j = t+1. Three rotating diagonal buffers in LDS. gamma = 1.
// Boundary "inf" stored directly as 1e10 (matches reference's inf->1e10
// substitution inside softmin; interior R values stay far below 1e10).
// ---------------------------------------------------------------------------
__global__ void softdtw_dp_kernel(const float* __restrict__ Dm,
                                  float* __restrict__ out) {
  const int b = blockIdx.x;
  const int t = threadIdx.x;           // 0..NN-1, j = t+1
  __shared__ float s0[NN + 2];
  __shared__ float s1[NN + 2];
  __shared__ float s2[NN + 2];
  float* d2r = s0;   // diag s-2
  float* d1r = s1;   // diag s-1
  float* cur = s2;   // diag s
  const float INFR = 1e10f;

  if (t == 0) {
    s0[0] = 0.0f;    // R[0][0]
    s1[0] = INFR;    // R[1][0]
    s1[1] = INFR;    // R[0][1]
  }
  __syncthreads();

  const float* Db = Dm + (size_t)b * NN * NN;

  for (int s = 2; s <= 2 * NN; ++s) {
    const int j = t + 1;
    const int i = s - j;
    if (i >= 1 && i <= NN) {
      float rdiag = d2r[j - 1];        // R[i-1][j-1]
      float rup   = d1r[j];            // R[i-1][j]
      float rleft = d1r[j - 1];        // R[i][j-1]
      float dv    = Db[(size_t)(i - 1) * NN + (j - 1)];
      float m   = fminf(rdiag, fminf(rup, rleft));
      float lse = __expf(m - rdiag) + __expf(m - rup) + __expf(m - rleft);
      cur[j] = dv + m - __logf(lse);   // gamma = 1
    }
    if (t == 0) {
      cur[0] = INFR;                   // R[s][0]
      if (s <= NN) cur[s] = INFR;      // R[0][s]
    }
    __syncthreads();
    if (s == 2 * NN && t == 0) out[b] = cur[NN];   // R[N][N]
    float* tmp = d2r; d2r = d1r; d1r = cur; cur = tmp;
  }
}

// ---------------------------------------------------------------------------
extern "C" void kernel_launch(void* const* d_in, const int* in_sizes, int n_in,
                              void* d_out, int out_size, void* d_ws, size_t ws_size,
                              hipStream_t stream) {
  const float* X = (const float*)d_in[0];
  const float* Y = (const float*)d_in[1];
  float* out = (float*)d_out;

  char* ws = (char*)d_ws;
  const size_t XH_OFF  = 0;
  const size_t YH_OFF  = XH_OFF + (size_t)NB * NN * ND * sizeof(_Float16); // 4 MB
  const size_t XSQ_OFF = YH_OFF + (size_t)NB * NN * ND * sizeof(_Float16); // 8 MB
  const size_t YSQ_OFF = XSQ_OFF + (size_t)NB * NN * sizeof(float);
  const size_t D_OFF   = YSQ_OFF + (size_t)NB * NN * sizeof(float);

  _Float16* Xh  = (_Float16*)(ws + XH_OFF);
  _Float16* Yh  = (_Float16*)(ws + YH_OFF);
  float*    xsq = (float*)(ws + XSQ_OFF);
  float*    ysq = (float*)(ws + YSQ_OFF);
  float*    Dm  = (float*)(ws + D_OFF);

  // 1) convert + norms: 2*NB*NN rows, one thread each
  {
    int threads = 256;
    int rows = 2 * NB * NN;
    softdtw_prep_kernel<<<rows / threads, threads, 0, stream>>>(X, Y, Xh, Yh, xsq, ysq);
  }
  // 2) WMMA distance GEMM: NB*32*32 tiles, 8 waves (tiles) per 256-thread block
  {
    int tiles = NB * 32 * 32;
    softdtw_gemm_dist_kernel<<<tiles / 8, 256, 0, stream>>>(Xh, Yh, xsq, ysq, Dm);
  }
  // 3) DP wavefront: one block per batch
  softdtw_dp_kernel<<<NB, NN, 0, stream>>>(Dm, out);
}
